// EncoderDecoderLSTM_82514911690879
// MI455X (gfx1250) — matrix operand order repaired and checked by reading, not compiled, not run on hardware
//
#include <hip/hip_runtime.h>
#include <hip/hip_bf16.h>

// ---------------------------------------------------------------------------
// EncoderDecoder LSTM for MI455X (gfx1250), wave32 + WMMA bf16.
//
//  * All weights converted once per call to bf16, pre-swizzled into the exact
//    V_WMMA_F32_16X16X32_BF16 A-fragment (weights) / B-fragment (lin_W) lane
//    layouts, so every lane issues one contiguous 32B load per fragment.
//    Packed weights ~70MB -> resident in the 192MB L2.
//  * Persistent kernel, 64 workgroups x 256 threads (8 waves). WG `wg` owns
//    h-slice [wg*16, wg*16+16) and the matching 64 gate rows (4 gates x 16).
//    Cell state c lives in LDS of the owning WG for the whole sequence.
//    Per layer: 8 waves = 4 gates x 2 K-halves accumulate WMMA partials with
//    a depth-3 software pipeline (3 A-fragment loads in flight per wave to
//    hide the L2 round-trip); LDS reduce; 16 threads do i/f/g/o + c/h update
//    and publish h (bf16) to a global ping-pong buffer staged back into LDS
//    with GLOBAL_LOAD_ASYNC_TO_LDS (ASYNCcnt). 2 grid barriers/step.
//  * Decoder layer-1 h is scatter-stored directly into WMMA A-fragment
//    layout; final logits = WMMA GEMM (255x1024 @ 1024x8192) + bias.
//  * Workspace requirement: ~72.5 MB.
// ---------------------------------------------------------------------------

#define Tt   128
#define NSs  128
#define VBv  128
#define Hh   1024
#define EMBe 512
#define VOCv 8192
#define TLEN 256
#define SEQ  (Tt * NSs)   // 16384
#define NWG  64

typedef __attribute__((ext_vector_type(16))) __bf16 v16bf;
typedef __attribute__((ext_vector_type(8)))  float  v8f;

__device__ __forceinline__ unsigned short f2bf(float f) {
  unsigned u = __float_as_uint(f);
  unsigned r = u + 0x7FFFu + ((u >> 16) & 1u);   // round-to-nearest-even
  return (unsigned short)(r >> 16);
}

__device__ __forceinline__ v8f wmma_bf16(v16bf a, v16bf b, v8f c) {
  return __builtin_amdgcn_wmma_f32_16x16x32_bf16(
      /*neg_a=*/false, a, /*neg_b=*/false, b,
      /*c_mod=*/(short)0, c, /*reuse_a=*/false, /*reuse_b=*/false);
}

// --------------------------- utility kernels -------------------------------

__global__ void zero_u32(unsigned* p, int n) {
  int i = blockIdx.x * 256 + threadIdx.x;
  if (i < n) p[i] = 0u;
}

// Pack f32 weight (R x C, row-major) into bf16 WMMA *A-fragment* tile order.
// Tile (rt,kt) = rows [rt*16,+16) x K [kt*32,+32); 512 bf16 per tile stored
// lane-major. Contiguous k order used here (same as B):
//   M = lane&15 ;  K = e + ((lane>=16) ? 16 : 0)
__global__ void pack_a(const float* __restrict__ W, unsigned short* __restrict__ dst,
                       int R, int C) {
  int idx = blockIdx.x * 256 + threadIdx.x;
  if (idx >= R * C) return;
  int tile = idx >> 9, w = idx & 511, lane = w >> 4, e = w & 15;
  int nkt = C >> 5;
  int rt = tile / nkt, kt = tile - rt * nkt;
  int m = lane & 15;
  int k = e + ((lane >> 4) << 4);
  dst[idx] = f2bf(W[(size_t)(rt * 16 + m) * C + (kt * 32 + k)]);
}

// Pack lin_W (8192 x 1024) into bf16 WMMA *B-fragment* tiles of lin_W^T.
// Contiguous k order used here (same as A): N = lane&15 ; K = e + ((lane>=16) ? 16 : 0)
__global__ void pack_b(const float* __restrict__ W, unsigned short* __restrict__ dst,
                       int R, int C) {
  int idx = blockIdx.x * 256 + threadIdx.x;
  if (idx >= R * C) return;
  int tile = idx >> 9, w = idx & 511, lane = w >> 4, e = w & 15;
  int nkt = C >> 5;
  int nt = tile / nkt, kt = tile - nt * nkt;
  int n = lane & 15;
  int kk = e + ((lane >> 4) << 4);
  dst[idx] = f2bf(W[(size_t)(nt * 16 + n) * C + (kt * 32 + kk)]);
}

// --------------------------- persistent LSTM -------------------------------

__device__ __forceinline__ void gbar(unsigned* cnt, unsigned target) {
  __syncthreads();
  if (threadIdx.x == 0) {
    __threadfence();
    __hip_atomic_fetch_add(cnt, 1u, __ATOMIC_RELEASE, __HIP_MEMORY_SCOPE_AGENT);
    while (__hip_atomic_load(cnt, __ATOMIC_ACQUIRE, __HIP_MEMORY_SCOPE_AGENT) < target)
      __builtin_amdgcn_s_sleep(2);
  }
  __syncthreads();
}

// Stage 2KB of global bf16 h-state into LDS with the gfx1250 async
// global->LDS DMA path (no VGPR data, tracked with ASYNCcnt).
// For flat LDS-aperture pointers, addr[31:0] is the LDS byte address.
__device__ __forceinline__ void copy_h_async(unsigned short* dst,
                                             const unsigned short* __restrict__ src) {
  unsigned lds = (unsigned)(uintptr_t)dst + threadIdx.x * 8u;
  const unsigned short* g = src + threadIdx.x * 4;   // 8 bytes per thread
  asm volatile("global_load_async_to_lds_b64 %0, %1, off"
               :: "v"(lds), "v"(g) : "memory");
}
__device__ __forceinline__ void async_wait() {
  asm volatile("s_wait_asynccnt 0x0" ::: "memory");
}

// Depth-3 software-pipelined fragment-GEMV accumulation: keeps 3 A-fragment
// global loads in flight per wave so the L2 latency is hidden behind WMMAs.
// Fully unrolled -> the rotation is pure register renaming.
template <int NIT>
__device__ __forceinline__ void accum_tiles(v8f& acc, const v16bf* __restrict__ W,
                                            size_t tbase, int lane,
                                            const unsigned short* vb, int kt0,
                                            int hioff) {
  if constexpr (NIT >= 3) {
    v16bf a0 = W[(tbase + 0) * 32 + lane];
    v16bf a1 = W[(tbase + 1) * 32 + lane];
    v16bf a2 = W[(tbase + 2) * 32 + lane];
#pragma unroll
    for (int i = 0; i < NIT - 3; ++i) {
      v16bf an = W[(tbase + i + 3) * 32 + lane];
      v16bf b  = *(const v16bf*)(vb + (kt0 + i) * 32 + hioff);
      acc = wmma_bf16(a0, b, acc);
      a0 = a1; a1 = a2; a2 = an;
    }
#pragma unroll
    for (int i = NIT - 3; i < NIT; ++i) {
      v16bf b = *(const v16bf*)(vb + (kt0 + i) * 32 + hioff);
      acc = wmma_bf16(a0, b, acc);
      a0 = a1; a1 = a2;
    }
  } else {
#pragma unroll
    for (int i = 0; i < NIT; ++i) {
      v16bf a = W[(tbase + i) * 32 + lane];
      v16bf b = *(const v16bf*)(vb + (kt0 + i) * 32 + hioff);
      acc = wmma_bf16(a, b, acc);
    }
  }
}

// One LSTM layer step for this WG's 16 h-elements.
// gates = x @ Wih^T + h_prev @ Whh^T + b ; c = f*c + i*g ; h = o*tanh(c)
template <int NKTX>
__device__ void lstm_layer(const unsigned short* __restrict__ wih,   // packed A
                           const unsigned short* __restrict__ whh,   // packed A
                           const unsigned short* xb,                 // LDS bf16[32*NKTX]
                           const unsigned short* hb,                 // LDS bf16[1024]
                           const float* __restrict__ bias,           // f32 [4096]
                           float* partial,                           // LDS [4][2][16]
                           float* cslice,                            // LDS [16]
                           unsigned short* __restrict__ gh_out,      // global bf16[1024]
                           unsigned short* __restrict__ outsA,       // optional A-pack
                           int srow, int wg) {
  const int tid = threadIdx.x, lane = tid & 31, wave = tid >> 5;
  const int g = wave & 3, kh = wave >> 2;           // gate, K-half
  const int rt = g * 64 + wg;                       // 16-row gate tile
  const v16bf* WA = (const v16bf*)wih;
  const v16bf* WH = (const v16bf*)whh;
  const int hioff = (lane >> 4) << 4;               // B broadcast lane-group
  v8f acc = {0.f, 0.f, 0.f, 0.f, 0.f, 0.f, 0.f, 0.f};

  // keep this wave's Whh stripe warm near the WGP
  __builtin_prefetch((const void*)(whh + ((size_t)(rt * 32 + kh * 16) << 9)), 0, 3);

  // x contribution
  {
    const int kt0 = kh * (NKTX / 2);
    accum_tiles<NKTX / 2>(acc, WA, (size_t)rt * NKTX + kt0, lane, xb, kt0, hioff);
  }
  // h contribution (K=1024 -> 16 pipelined WMMAs per wave)
  {
    const int kt0 = kh * 16;
    accum_tiles<16>(acc, WH, (size_t)rt * 32 + kt0, lane, hb, kt0, hioff);
  }
  // All B columns identical -> every lane holds the 8 row-results for its
  // half: lanes 0-15 rows M=r, lanes 16-31 rows M=r+8.
  if (lane == 0 || lane == 16) {
    float* p = partial + (g * 2 + kh) * 16 + ((lane >> 4) << 3);
#pragma unroll
    for (int r = 0; r < 8; ++r) p[r] = acc[r];
  }
  __syncthreads();
  if (tid < 16) {
    const int m = tid, j = wg * 16 + m;
    float gi = partial[(0 * 2 + 0) * 16 + m] + partial[(0 * 2 + 1) * 16 + m] + bias[0 * Hh + j];
    float gf = partial[(1 * 2 + 0) * 16 + m] + partial[(1 * 2 + 1) * 16 + m] + bias[1 * Hh + j];
    float gg = partial[(2 * 2 + 0) * 16 + m] + partial[(2 * 2 + 1) * 16 + m] + bias[2 * Hh + j];
    float go = partial[(3 * 2 + 0) * 16 + m] + partial[(3 * 2 + 1) * 16 + m] + bias[3 * Hh + j];
    float i_ = 1.f / (1.f + __expf(-gi));
    float f_ = 1.f / (1.f + __expf(-gf));
    float g_ = tanhf(gg);
    float o_ = 1.f / (1.f + __expf(-go));
    float c  = f_ * cslice[m] + i_ * g_;
    cslice[m] = c;
    float h = o_ * tanhf(c);
    unsigned short hb16 = f2bf(h);
    gh_out[j] = hb16;
    if (outsA) {  // scatter into WMMA A-fragment layout (rows = timesteps)
      int rt2 = srow >> 4, mr = srow & 15, kt2 = j >> 5, k = j & 31;
      int e = k & 15, hi = k >> 4;
      int lanep = hi * 16 + mr;
      outsA[((size_t)(rt2 * 32 + kt2) << 9) + (lanep << 4) + e] = hb16;
    }
  }
  __syncthreads();
}

__global__ __launch_bounds__(256, 1) void lstm_persistent(
    const float* __restrict__ source, const int* __restrict__ target,
    const float* __restrict__ dec_emb,
    const float* __restrict__ enc_b0, const float* __restrict__ enc_b1,
    const float* __restrict__ dec_b0, const float* __restrict__ dec_b1,
    const unsigned short* __restrict__ eWih0, const unsigned short* __restrict__ eWhh0,
    const unsigned short* __restrict__ eWih1, const unsigned short* __restrict__ eWhh1,
    const unsigned short* __restrict__ dWih0, const unsigned short* __restrict__ dWhh0,
    const unsigned short* __restrict__ dWih1, const unsigned short* __restrict__ dWhh1,
    unsigned short* gh,          // [4][1024] : gh0{p0,p1}, gh1{p0,p1} (bf16)
    unsigned short* outsA,       // decoder outs, A-fragment packed (256x1024)
    unsigned* barrier_cnt) {
  __shared__ alignas(32) unsigned short xb[EMBe];
  __shared__ alignas(32) unsigned short hb0[Hh];
  __shared__ alignas(32) unsigned short hb1[Hh];
  __shared__ float partial[128];
  __shared__ float cs0[16], cs1[16];
  const int tid = threadIdx.x, wg = blockIdx.x;
  if (tid < 16) { cs0[tid] = 0.f; cs1[tid] = 0.f; }
  unsigned gen = 0;
  unsigned short* gh0 = gh;
  unsigned short* gh1 = gh + 2 * Hh;

  // ---------------- encoder: 16384 sequential steps ----------------
  for (int s = 0; s < SEQ; ++s) {
    const int cur = s & 1, prev = cur ^ 1;
    // xs[s] = source[s%T, s/T, :]
    const float* x = source + (size_t)((s & (Tt - 1)) * NSs + (s >> 7)) * VBv;
    copy_h_async(hb0, gh0 + prev * Hh);
    if (tid < VBv) xb[tid] = f2bf(x[tid]);
    async_wait();
    __syncthreads();
    lstm_layer<VBv / 32>(eWih0, eWhh0, xb, hb0, enc_b0, partial, cs0,
                         gh0 + cur * Hh, nullptr, 0, wg);
    gbar(barrier_cnt, ++gen);
    copy_h_async(hb0, gh0 + cur * Hh);
    copy_h_async(hb1, gh1 + prev * Hh);
    async_wait();
    __syncthreads();
    lstm_layer<Hh / 32>(eWih1, eWhh1, hb0, hb1, enc_b1, partial, cs1,
                        gh1 + cur * Hh, nullptr, 0, wg);
    gbar(barrier_cnt, ++gen);
  }

  // ---------------- decoder: 255 steps (h,c carried over) ----------------
  for (int d = 0; d < TLEN - 1; ++d) {
    const int cur = d & 1, prev = cur ^ 1;  // SEQ even -> parity continues
    const float* x = dec_emb + (size_t)target[d] * EMBe;
    copy_h_async(hb0, gh0 + prev * Hh);
    { float2 v = ((const float2*)x)[tid]; xb[tid * 2] = f2bf(v.x); xb[tid * 2 + 1] = f2bf(v.y); }
    async_wait();
    __syncthreads();
    lstm_layer<EMBe / 32>(dWih0, dWhh0, xb, hb0, dec_b0, partial, cs0,
                          gh0 + cur * Hh, nullptr, 0, wg);
    gbar(barrier_cnt, ++gen);
    copy_h_async(hb0, gh0 + cur * Hh);
    copy_h_async(hb1, gh1 + prev * Hh);
    async_wait();
    __syncthreads();
    lstm_layer<Hh / 32>(dWih1, dWhh1, hb0, hb1, dec_b1, partial, cs1,
                        gh1 + cur * Hh, outsA, d, wg);
    gbar(barrier_cnt, ++gen);
  }
}

// --------------------------- logits WMMA GEMM ------------------------------
// out[1+s, :] = outs[s] @ lin_W^T + lin_b   (255x1024 @ 1024x8192)
__global__ __launch_bounds__(256) void logits_gemm(
    const unsigned short* __restrict__ outsA,   // A-packed (256x1024)
    const unsigned short* __restrict__ linWB,   // B-packed (1024x8192 as B tiles)
    const float* __restrict__ lin_b, float* __restrict__ out) {
  const int gwave = (blockIdx.x * 256 + threadIdx.x) >> 5;  // 0..8191
  const int lane = threadIdx.x & 31;
  const int mt = gwave >> 9;          // 16 row tiles
  const int nt = gwave & 511;         // 512 col tiles
  const v16bf* A = (const v16bf*)outsA;
  const v16bf* B = (const v16bf*)linWB;
  v8f acc = {0.f, 0.f, 0.f, 0.f, 0.f, 0.f, 0.f, 0.f};
  // depth-3 pipeline: 3 (a,b) global-load pairs in flight
  v16bf a0 = A[(size_t)(mt * 32 + 0) * 32 + lane];
  v16bf b0 = B[(size_t)(nt * 32 + 0) * 32 + lane];
  v16bf a1 = A[(size_t)(mt * 32 + 1) * 32 + lane];
  v16bf b1 = B[(size_t)(nt * 32 + 1) * 32 + lane];
  v16bf a2 = A[(size_t)(mt * 32 + 2) * 32 + lane];
  v16bf b2 = B[(size_t)(nt * 32 + 2) * 32 + lane];
#pragma unroll
  for (int kt = 0; kt < 29; ++kt) {
    v16bf an = A[(size_t)(mt * 32 + kt + 3) * 32 + lane];
    v16bf bn = B[(size_t)(nt * 32 + kt + 3) * 32 + lane];
    acc = wmma_bf16(a0, b0, acc);
    a0 = a1; a1 = a2; a2 = an;
    b0 = b1; b1 = b2; b2 = bn;
  }
  acc = wmma_bf16(a0, b0, acc);
  acc = wmma_bf16(a1, b1, acc);
  acc = wmma_bf16(a2, b2, acc);
  const int n = nt * 16 + (lane & 15);
  const int rbase = mt * 16 + ((lane >> 4) << 3);
  const float bias = lin_b[n];
#pragma unroll
  for (int r = 0; r < 8; ++r) {
    int srow = rbase + r;
    if (srow < TLEN - 1) out[(size_t)(srow + 1) * VOCv + n] = acc[r] + bias;
  }
}

// ------------------------------ launcher -----------------------------------

extern "C" void kernel_launch(void* const* d_in, const int* in_sizes, int n_in,
                              void* d_out, int out_size, void* d_ws, size_t ws_size,
                              hipStream_t stream) {
  (void)in_sizes; (void)n_in; (void)out_size; (void)ws_size;
  const float* source  = (const float*)d_in[0];
  const int*   target  = (const int*)  d_in[1];
  const float* eWih0f  = (const float*)d_in[2];
  const float* eWhh0f  = (const float*)d_in[3];
  const float* enc_b0  = (const float*)d_in[4];
  const float* eWih1f  = (const float*)d_in[5];
  const float* eWhh1f  = (const float*)d_in[6];
  const float* enc_b1  = (const float*)d_in[7];
  const float* dec_emb = (const float*)d_in[8];
  const float* dWih0f  = (const float*)d_in[9];
  const float* dWhh0f  = (const float*)d_in[10];
  const float* dec_b0  = (const float*)d_in[11];
  const float* dWih1f  = (const float*)d_in[12];
  const float* dWhh1f  = (const float*)d_in[13];
  const float* dec_b1  = (const float*)d_in[14];
  const float* linWf   = (const float*)d_in[15];
  const float* lin_b   = (const float*)d_in[16];
  float* out = (float*)d_out;

  // Workspace layout (1KB aligned). Total ~72.5 MB.
  char* ws = (char*)d_ws;
  size_t off = 0;
  auto take = [&](size_t bytes) {
    size_t o = off; off += (bytes + 1023) & ~(size_t)1023; return o;
  };
  const size_t o_eWih0 = take((size_t)4096 * 128  * 2);
  const size_t o_eWhh0 = take((size_t)4096 * 1024 * 2);
  const size_t o_eWih1 = take((size_t)4096 * 1024 * 2);
  const size_t o_eWhh1 = take((size_t)4096 * 1024 * 2);
  const size_t o_dWih0 = take((size_t)4096 * 512  * 2);
  const size_t o_dWhh0 = take((size_t)4096 * 1024 * 2);
  const size_t o_dWih1 = take((size_t)4096 * 1024 * 2);
  const size_t o_dWhh1 = take((size_t)4096 * 1024 * 2);
  const size_t o_linB  = take((size_t)8192 * 1024 * 2);
  const size_t o_gh    = take((size_t)4 * 1024 * 2);      // state region start
  const size_t o_outsA = take((size_t)256 * 1024 * 2);
  const size_t o_cnt   = take(4);
  const size_t state_bytes = off - o_gh;

  // 1) zero output and state (barrier counter, h ping-pong, outs padding)
  {
    int n = (TLEN * VOCv);  // 2,097,152 f32
    zero_u32<<<(n + 255) / 256, 256, 0, stream>>>((unsigned*)out, n);
    int m = (int)(state_bytes / 4);
    zero_u32<<<(m + 255) / 256, 256, 0, stream>>>((unsigned*)(ws + o_gh), m);
  }
  // 2) pack weights f32 -> bf16 WMMA fragments (L2-resident afterwards)
  auto launch_pa = [&](const float* W, size_t o, int R, int C) {
    int n = R * C;
    pack_a<<<(n + 255) / 256, 256, 0, stream>>>(W, (unsigned short*)(ws + o), R, C);
  };
  launch_pa(eWih0f, o_eWih0, 4096, 128);
  launch_pa(eWhh0f, o_eWhh0, 4096, 1024);
  launch_pa(eWih1f, o_eWih1, 4096, 1024);
  launch_pa(eWhh1f, o_eWhh1, 4096, 1024);
  launch_pa(dWih0f, o_dWih0, 4096, 512);
  launch_pa(dWhh0f, o_dWhh0, 4096, 1024);
  launch_pa(dWih1f, o_dWih1, 4096, 1024);
  launch_pa(dWhh1f, o_dWhh1, 4096, 1024);
  {
    int n = 8192 * 1024;
    pack_b<<<(n + 255) / 256, 256, 0, stream>>>(linWf, (unsigned short*)(ws + o_linB),
                                                8192, 1024);
  }
  // 3) persistent recurrent kernel (encoder + decoder)
  lstm_persistent<<<NWG, 256, 0, stream>>>(
      source, target, dec_emb, enc_b0, enc_b1, dec_b0, dec_b1,
      (const unsigned short*)(ws + o_eWih0), (const unsigned short*)(ws + o_eWhh0),
      (const unsigned short*)(ws + o_eWih1), (const unsigned short*)(ws + o_eWhh1),
      (const unsigned short*)(ws + o_dWih0), (const unsigned short*)(ws + o_dWhh0),
      (const unsigned short*)(ws + o_dWih1), (const unsigned short*)(ws + o_dWhh1),
      (unsigned short*)(ws + o_gh), (unsigned short*)(ws + o_outsA),
      (unsigned*)(ws + o_cnt));
  // 4) logits GEMM: 8192 WMMA tiles, one per wave
  logits_gemm<<<1024, 256, 0, stream>>>((const unsigned short*)(ws + o_outsA),
                                        (const unsigned short*)(ws + o_linB),
                                        lin_b, out);
}
